// MultiHeadAttention_90615220011189
// MI455X (gfx1250) — matrix operand order repaired and checked by reading, not compiled, not run on hardware
//
#include <hip/hip_runtime.h>

#define B_ 4
#define N_ 4096
#define U_ 256
#define H_ 4
#define DH_ 64

#define LOG2E 1.4426950408889634f
#define NEG_MASK_LOG2E (-10000.0f * LOG2E)

typedef __attribute__((ext_vector_type(2)))  float    v2f;
typedef __attribute__((ext_vector_type(8)))  float    v8f;
typedef __attribute__((ext_vector_type(2)))  _Float16 v2h;
typedef __attribute__((ext_vector_type(4)))  _Float16 v4h;
typedef __attribute__((ext_vector_type(8)))  _Float16 v8h;
typedef __attribute__((ext_vector_type(16))) _Float16 v16h;
typedef __attribute__((ext_vector_type(2)))  __fp16   v2hf;

// fp32 WMMA: D = A(16x4) x B(4x16) + C  (exact projections)
__device__ __forceinline__ v8f wmma_f32(v2f a, v2f b, v8f c) {
  return __builtin_amdgcn_wmma_f32_16x16x4_f32(
      false, a, false, b, (short)0, c, false, false);
}

// f16 WMMA: D = A(16x32) x B(32x16) + C, fp32 accumulate (attention core)
__device__ __forceinline__ v8f wmma_f16(v16h a, v16h b, v8f c) {
  return __builtin_amdgcn_wmma_f32_16x16x32_f16(
      false, a, false, b, (short)0, c, false, false);
}

// lane l <-> lane l^16 exchange (group-of-32 xor swizzle, SWAPX16)
__device__ __forceinline__ float swapx16(float x) {
  return __int_as_float(__builtin_amdgcn_ds_swizzle(__float_as_int(x), 0x401F));
}

__device__ __forceinline__ v16h cat8(v8h a, v8h b) {
  return __builtin_shufflevector(a, b, 0, 1, 2, 3, 4, 5, 6, 7,
                                 8, 9, 10, 11, 12, 13, 14, 15);
}

__device__ __forceinline__ v2h pkrtz(float a, float b) {
  v2hf r = __builtin_amdgcn_cvt_pkrtz(a, b);
  return __builtin_bit_cast(v2h, r);
}

// ---------------------------------------------------------------------------
// Projection: Y = x @ W.T in exact fp32 WMMA, stored as f16.
// z=0 -> Q*(scale*log2e) [b][h][n][64], z=1 -> K [b][h][n][64],
// z=2 -> V^T [b][h][64][n]
// ---------------------------------------------------------------------------
__global__ void __launch_bounds__(256)
proj_kernel(const float* __restrict__ x,
            const float* __restrict__ Wq,
            const float* __restrict__ Wk,
            const float* __restrict__ Wv,
            _Float16* __restrict__ Qh,
            _Float16* __restrict__ Kh,
            _Float16* __restrict__ VTh) {
  const int lane = threadIdx.x;
  const int wave = threadIdx.y;
  const int llo = lane & 15, lhi = lane >> 4;
  const int r0 = (blockIdx.x * 8 + wave) * 16;
  const int c0 = blockIdx.y * 16;
  const int z = blockIdx.z;
  const float* W = (z == 0) ? Wq : (z == 1) ? Wk : Wv;

  const float* ap = x + (size_t)(r0 + llo) * U_ + 2 * lhi;
  const float* bp = W + (size_t)(c0 + llo) * U_ + 2 * lhi;

  v8f acc = {0.f, 0.f, 0.f, 0.f, 0.f, 0.f, 0.f, 0.f};
#pragma unroll 8
  for (int s = 0; s < 64; ++s) {
    v2f a = *(const v2f*)(ap + 4 * s);
    v2f b = *(const v2f*)(bp + 4 * s);
    acc = wmma_f32(a, b, acc);
  }

  // softmax scale (1/sqrt(64)) and log2e folded into Q
  const float scale = (z == 0) ? (0.125f * LOG2E) : 1.0f;
  const int bidx = r0 / N_;
  const int n0 = (r0 % N_) + 8 * lhi;
  const int h = c0 >> 6;
  const int d = (c0 & 63) + llo;
  if (z < 2) {
    _Float16* base = ((z == 0) ? Qh : Kh) + (size_t)(bidx * H_ + h) * N_ * DH_;
#pragma unroll
    for (int v = 0; v < 8; ++v)
      base[(size_t)(n0 + v) * DH_ + d] = (_Float16)(acc[v] * scale);
  } else {
    _Float16* base = VTh + ((size_t)(bidx * H_ + h) * DH_ + d) * N_;
#pragma unroll
    for (int v = 0; v < 8; ++v)
      base[n0 + v] = (_Float16)acc[v];
  }
}

// ---------------------------------------------------------------------------
// One 32-key tile of online-softmax attention (log2 domain).
// ---------------------------------------------------------------------------
template <bool USE_MASK>
__device__ __forceinline__ void attn_tile(
    int j0, int qi, int llo, int lhi, bool lo,
    const _Float16* __restrict__ Kb, const _Float16* __restrict__ VTb,
    const v16h* __restrict__ qf, v8f* __restrict__ acc,
    float& m, float& lsum) {
  const v8f vzero = {0.f, 0.f, 0.f, 0.f, 0.f, 0.f, 0.f, 0.f};

  // ---- S^T = K_tile x Q^T : two 16-key sub-tiles, K-dim 64 = 2 steps ----
  v8f st0 = vzero, st1 = vzero;
  {
    const _Float16* kp = Kb + (size_t)(j0 + llo) * DH_;
    v16h a0 = cat8(*(const v8h*)(kp + 8 * lhi),
                   *(const v8h*)(kp + 16 + 8 * lhi));
    v16h a1 = cat8(*(const v8h*)(kp + 32 + 8 * lhi),
                   *(const v8h*)(kp + 48 + 8 * lhi));
    st0 = wmma_f16(a0, qf[0], st0);
    st0 = wmma_f16(a1, qf[1], st0);
  }
  {
    const _Float16* kp = Kb + (size_t)(j0 + 16 + llo) * DH_;
    v16h a0 = cat8(*(const v8h*)(kp + 8 * lhi),
                   *(const v8h*)(kp + 16 + 8 * lhi));
    v16h a1 = cat8(*(const v8h*)(kp + 32 + 8 * lhi),
                   *(const v8h*)(kp + 48 + 8 * lhi));
    st1 = wmma_f16(a0, qf[0], st1);
    st1 = wmma_f16(a1, qf[1], st1);
  }

  if (USE_MASK) {
    // +(-10000*log2e) where key j <= query i
#pragma unroll
    for (int v = 0; v < 8; ++v) {
      const int kg0 = j0 + v + 8 * lhi;
      st0[v] += (kg0 > qi) ? 0.f : NEG_MASK_LOG2E;
      st1[v] += (kg0 + 16 > qi) ? 0.f : NEG_MASK_LOG2E;
    }
  }

  // ---- online softmax over the 32-key tile (per query q = llo) ----
  float mt;
  {
    float a0 = fmaxf(st0[0], st0[1]), a1 = fmaxf(st0[2], st0[3]);
    float a2 = fmaxf(st0[4], st0[5]), a3 = fmaxf(st0[6], st0[7]);
    float b0 = fmaxf(st1[0], st1[1]), b1 = fmaxf(st1[2], st1[3]);
    float b2 = fmaxf(st1[4], st1[5]), b3 = fmaxf(st1[6], st1[7]);
    mt = fmaxf(fmaxf(fmaxf(a0, a1), fmaxf(a2, a3)),
               fmaxf(fmaxf(b0, b1), fmaxf(b2, b3)));
  }
  mt = fmaxf(mt, swapx16(mt));
  const float mnew = fmaxf(m, mt);
  const float c = __builtin_amdgcn_exp2f(m - mnew);
  float p0[8], p1[8];
#pragma unroll
  for (int v = 0; v < 8; ++v) {
    p0[v] = __builtin_amdgcn_exp2f(st0[v] - mnew);
    p1[v] = __builtin_amdgcn_exp2f(st1[v] - mnew);
  }
  float ls;
  {
    float a0 = (p0[0] + p0[1]) + (p0[2] + p0[3]);
    float a1 = (p0[4] + p0[5]) + (p0[6] + p0[7]);
    float b0 = (p1[0] + p1[1]) + (p1[2] + p1[3]);
    float b1 = (p1[4] + p1[5]) + (p1[6] + p1[7]);
    ls = (a0 + a1) + (b0 + b1);
  }
  ls += swapx16(ls);
  lsum = lsum * c + ls;
  m = mnew;
#pragma unroll
  for (int t = 0; t < 4; ++t) acc[t] *= c;

  // ---- rewire P (two S^T C-tiles) into the P^T B-fragment (f16) ----
  float sw0[8], sw1[8];
#pragma unroll
  for (int v = 0; v < 8; ++v) {
    sw0[v] = swapx16(p0[v]);
    sw1[v] = swapx16(p1[v]);
  }
  // element pair j of bf: K = 16*lhi + 2j, 2j+1
  v2h e0 = lo ? pkrtz(p0[0], p0[1]) : pkrtz(sw1[0], sw1[1]);
  v2h e1 = lo ? pkrtz(p0[2], p0[3]) : pkrtz(sw1[2], sw1[3]);
  v2h e2 = lo ? pkrtz(p0[4], p0[5]) : pkrtz(sw1[4], sw1[5]);
  v2h e3 = lo ? pkrtz(p0[6], p0[7]) : pkrtz(sw1[6], sw1[7]);
  v2h e4 = lo ? pkrtz(sw0[0], sw0[1]) : pkrtz(p1[0], p1[1]);
  v2h e5 = lo ? pkrtz(sw0[2], sw0[3]) : pkrtz(p1[2], p1[3]);
  v2h e6 = lo ? pkrtz(sw0[4], sw0[5]) : pkrtz(p1[4], p1[5]);
  v2h e7 = lo ? pkrtz(sw0[6], sw0[7]) : pkrtz(p1[6], p1[7]);
  v4h q0h = __builtin_shufflevector(e0, e1, 0, 1, 2, 3);
  v4h q1h = __builtin_shufflevector(e2, e3, 0, 1, 2, 3);
  v4h q2h = __builtin_shufflevector(e4, e5, 0, 1, 2, 3);
  v4h q3h = __builtin_shufflevector(e6, e7, 0, 1, 2, 3);
  v8h h0 = __builtin_shufflevector(q0h, q1h, 0, 1, 2, 3, 4, 5, 6, 7);
  v8h h1 = __builtin_shufflevector(q2h, q3h, 0, 1, 2, 3, 4, 5, 6, 7);
  v16h bf = cat8(h0, h1);

  // ---- O^T += V^T_block x P^T : 4 dh-blocks, one K=32 WMMA each ----
#pragma unroll
  for (int t = 0; t < 4; ++t) {
    const _Float16* vp = VTb + (size_t)(16 * t + llo) * N_ + j0;
    v16h va = cat8(*(const v8h*)(vp + 16 * lhi),
                   *(const v8h*)(vp + 8 + 16 * lhi));
    acc[t] = wmma_f16(va, bf, acc[t]);
  }
}

// ---------------------------------------------------------------------------
// Flash attention (anti-causal -10000 mask), one 16-query tile per wave.
// Key tiles fully below the query tile are skipped (their softmax weights
// underflow to exactly 0 in fp32, same as the reference) — except the last
// query tile (row 4095 softmaxes over all-masked keys), which runs the full
// masked range.
// ---------------------------------------------------------------------------
__global__ void __launch_bounds__(256)
flash_attn_kernel(const _Float16* __restrict__ Qh,
                  const _Float16* __restrict__ Kh,
                  const _Float16* __restrict__ VTh,
                  const unsigned char* __restrict__ in_mask,
                  float* __restrict__ out) {
  const int lane = threadIdx.x;
  const int wave = threadIdx.y;
  const int llo = lane & 15, lhi = lane >> 4;
  const int bh = blockIdx.y;
  const int b = bh >> 2, h = bh & 3;
  const int q0 = (blockIdx.x * 8 + wave) * 16;

  const _Float16* Qb = Qh + (size_t)(b * H_ + h) * N_ * DH_;
  const _Float16* Kb = Kh + (size_t)(b * H_ + h) * N_ * DH_;
  const _Float16* VTb = VTh + (size_t)(b * H_ + h) * DH_ * N_;

  // Q^T B-fragments: chunk c covers d = 32c + 8*lhi + 0..7 and 32c + 16 + 8*lhi + 0..7
  v16h qf[2];
  {
    const _Float16* qp = Qb + (size_t)(q0 + llo) * DH_ + 8 * lhi;
    qf[0] = cat8(*(const v8h*)(qp), *(const v8h*)(qp + 16));
    qf[1] = cat8(*(const v8h*)(qp + 32), *(const v8h*)(qp + 48));
  }

  const v8f vzero = {0.f, 0.f, 0.f, 0.f, 0.f, 0.f, 0.f, 0.f};
  v8f acc[4];
#pragma unroll
  for (int t = 0; t < 4; ++t) acc[t] = vzero;

  float m = -3.0e38f;
  float lsum = 0.f;
  const int qi = q0 + llo;
  const bool lo = (lhi == 0);

  // first kept tile; last query tile must process everything (masked path)
  int j0 = (q0 == N_ - 16) ? 0 : (q0 & ~31);
  const int diag_end = q0 + 16;  // tiles with j0 < diag_end straddle the mask
  for (; j0 < diag_end; j0 += 32)
    attn_tile<true>(j0, qi, llo, lhi, lo, Kb, VTb, qf, acc, m, lsum);
  for (; j0 < N_; j0 += 32)
    attn_tile<false>(j0, qi, llo, lhi, lo, Kb, VTb, qf, acc, m, lsum);

  // finalize: divide by softmax denom, apply ~input_mask, store (b,n,u) fp32
  float r = 1.0f / lsum;
  if (in_mask[b * N_ + qi]) r = 0.f;
  float* ob = out + ((size_t)b * N_ + q0 + llo) * U_ + h * DH_;
#pragma unroll
  for (int t = 0; t < 4; ++t)
#pragma unroll
    for (int v = 0; v < 8; ++v)
      ob[16 * t + v + 8 * lhi] = acc[t][v] * r;
}

extern "C" void kernel_launch(void* const* d_in, const int* in_sizes, int n_in,
                              void* d_out, int out_size, void* d_ws, size_t ws_size,
                              hipStream_t stream) {
  (void)in_sizes; (void)n_in; (void)out_size; (void)ws_size;
  const float* x = (const float*)d_in[0];
  const unsigned char* msk = (const unsigned char*)d_in[1];
  const float* Wq = (const float*)d_in[2];
  const float* Wk = (const float*)d_in[3];
  const float* Wv = (const float*)d_in[4];
  float* out = (float*)d_out;

  const size_t mat_elems = (size_t)B_ * H_ * N_ * DH_;  // 4,194,304
  _Float16* Qh = (_Float16*)d_ws;
  _Float16* Kh = Qh + mat_elems;
  _Float16* VTh = Kh + mat_elems;

  dim3 pb(32, 8);
  dim3 pg(B_ * N_ / 16 / 8, U_ / 16, 3);  // 128 x 16 x 3
  proj_kernel<<<pg, pb, 0, stream>>>(x, Wq, Wk, Wv, Qh, Kh, VTh);

  dim3 ab(32, 8);
  dim3 ag(N_ / 16 / 8, B_ * H_);          // 32 x 16
  flash_attn_kernel<<<ag, ab, 0, stream>>>(Qh, Kh, VTh, msk, out);
}
